// GatedDeltaNet_68255620268362
// MI455X (gfx1250) — compile-verified, matrix operand order checked by reading, not run on hardware
//
#include <hip/hip_runtime.h>
#include <hip/hip_bf16.h>

// ---------------- problem constants ----------------
constexpr int B_  = 2;
constexpr int S_  = 2048;
constexpr int H_  = 2048;
constexpr int HK_ = 16;
constexpr int HV_ = 32;
constexpr int DK_ = 128;
constexpr int DV_ = 128;
constexpr int G_  = HV_ / HK_;          // 2
constexpr int K_  = 4;                  // conv taps
constexpr int CD_ = 2 * HK_ * DK_ + HV_ * DV_;   // 8192
constexpr int FQKVZ_ = CD_ + HV_ * DV_;          // 12288
constexpr int FBA_ = 2 * HV_;                    // 64
constexpr float EPS_ = 1e-6f;

typedef __attribute__((ext_vector_type(16))) __bf16          v16bf;
typedef __attribute__((ext_vector_type(8)))  float           v8f;
typedef __attribute__((ext_vector_type(16))) unsigned short  v16us;
typedef __attribute__((ext_vector_type(4)))  unsigned int    u32x4;
typedef __attribute__((ext_vector_type(8)))  int             i32x8;
typedef __attribute__((ext_vector_type(4)))  int             i32x4;

__device__ __forceinline__ unsigned short f32_to_bf16_rne(float f) {
    union { float f; unsigned int u; } v; v.f = f;
    unsigned int u = v.u;
    unsigned int r = u + 0x7FFFu + ((u >> 16) & 1u);
    return (unsigned short)(r >> 16);
}

#if __has_builtin(__builtin_amdgcn_tensor_load_to_lds)
#define HAVE_TDM 1
#else
#define HAVE_TDM 0
#endif

#if HAVE_TDM
// Issue a 2D TDM tile load (bf16 elements) global -> LDS with per-row padding.
// pad_code: LDS pad of 1 DWORD every 2^(pad_code+1) DWORDs stored.
__device__ __forceinline__ void tdm_load_2d_bf16(
    const void* gptr, unsigned lds_byte_addr,
    unsigned tensor_d0, unsigned tensor_d1,
    unsigned tile_d0, unsigned tile_d1,
    unsigned long stride0_elems, unsigned pad_code)
{
    u32x4 g0;
    i32x8 g1;
    i32x4 g2 = {0, 0, 0, 0};
    i32x4 g3 = {0, 0, 0, 0};
    const unsigned long ga = (unsigned long)gptr;
    g0[0] = 1u;                                   // count=1 (valid descriptor)
    g0[1] = lds_byte_addr;                        // lds_addr [63:32]
    g0[2] = (unsigned)(ga & 0xFFFFFFFFu);         // global_addr [95:64]
    g0[3] = (unsigned)((ga >> 32) & 0x1FFFFFFu)   // global_addr [120:96]
          | (2u << 30);                           // type=2 (image) [127:126]
    // group1: data_size=2B (code 1), pad_enable, pad_interval=pad_code, pad_amount=0
    g1[0] = (int)((1u << 16) | (1u << 20) | (pad_code << 22));
    g1[1] = (int)((tensor_d0 & 0xFFFFu) << 16);                    // tensor_dim0 lo
    g1[2] = (int)((tensor_d0 >> 16) | ((tensor_d1 & 0xFFFFu) << 16));
    g1[3] = (int)((tensor_d1 >> 16) | (tile_d0 << 16));            // tile_dim0
    g1[4] = (int)(tile_d1);                                        // tile_dim1 (tile_dim2=0)
    g1[5] = (int)(unsigned)(stride0_elems & 0xFFFFFFFFul);         // stride0 lo32
    g1[6] = (int)(unsigned)((stride0_elems >> 32) & 0xFFFFul);     // stride0 hi16
    g1[7] = 0;
#if __clang_major__ >= 23
    i32x8 g4 = {0, 0, 0, 0, 0, 0, 0, 0};
    __builtin_amdgcn_tensor_load_to_lds(g0, g1, g2, g3, g4, 0);
#else
    __builtin_amdgcn_tensor_load_to_lds(g0, g1, g2, g3, 0);
#endif
}

__device__ __forceinline__ void wait_tensorcnt0() {
#if __has_builtin(__builtin_amdgcn_s_wait_tensorcnt)
    __builtin_amdgcn_s_wait_tensorcnt(0);
#else
    asm volatile("s_wait_tensorcnt 0x0" ::: "memory");
#endif
}
#endif // HAVE_TDM

// ---------------------------------------------------------------------------
// f32 -> bf16 conversion pass
// ---------------------------------------------------------------------------
__global__ __launch_bounds__(256) void cvt_bf16_kernel(
    const float* __restrict__ in, unsigned short* __restrict__ out, long n)
{
    const long i = (long)blockIdx.x * 256 + threadIdx.x;
    if (i < n) out[i] = f32_to_bf16_rne(in[i]);
}

// ---------------------------------------------------------------------------
// WMMA bf16 GEMM with TDM double-buffered LDS staging.
// C(M,N) f32 = A(M,K) bf16 * B(K,N) bf16, row-major. M,N mult of 128, K of 32.
// ---------------------------------------------------------------------------
__global__ __launch_bounds__(256) void wmma_gemm_bf16_tdm(
    const unsigned short* __restrict__ A, const unsigned short* __restrict__ Bm,
    float* __restrict__ C, int M, int N, int K)
{
    constexpr int BM = 128, BN = 128, BK = 32;
    constexpr int LDA = BK + 2;    // ushorts: 64B row + 4B pad
    constexpr int LDB = BN + 2;    // ushorts: 256B row + 4B pad
    __shared__ unsigned short sA[2][BM][LDA];
    __shared__ unsigned short sB[2][BK][LDB];

    const int tid  = threadIdx.x;
    const int lane = tid & 31;
    const int w    = tid >> 5;
    const int waveRow = (w & 1) * 64;
    const int waveCol = (w >> 1) * 32;
    const int rowBase = blockIdx.y * BM;
    const int colBase = blockIdx.x * BN;

    v8f acc[4][2];
    #pragma unroll
    for (int i = 0; i < 4; ++i)
        #pragma unroll
        for (int j = 0; j < 2; ++j)
            acc[i][j] = (v8f){0.f,0.f,0.f,0.f,0.f,0.f,0.f,0.f};

    const int half = lane >> 4;
    const int mn   = lane & 15;
    const int nk   = K / BK;

#if HAVE_TDM
    // preload chunk 0 into buffer 0 (one DMA pair, issued by wave 0 / lane 0)
    if (tid == 0) {
        tdm_load_2d_bf16(A + (size_t)rowBase * K,
                         (unsigned)(size_t)&sA[0][0][0],
                         (unsigned)K, (unsigned)M, BK, BM, (unsigned long)K, 3u);
        tdm_load_2d_bf16(Bm + colBase,
                         (unsigned)(size_t)&sB[0][0][0],
                         (unsigned)N, (unsigned)K, BN, BK, (unsigned long)N, 5u);
        wait_tensorcnt0();
    }
    __syncthreads();
#else
    {   // synchronous staging of chunk 0
        const int la_r = tid >> 1, la_c = (tid & 1) * 16;
        const int lb_r = tid >> 3, lb_c = (tid & 7) * 16;
        const unsigned short* ap = A + (size_t)(rowBase + la_r) * K + la_c;
        const unsigned short* bp = Bm + (size_t)lb_r * N + colBase + lb_c;
        #pragma unroll
        for (int j = 0; j < 16; ++j) sA[0][la_r][la_c + j] = ap[j];
        #pragma unroll
        for (int j = 0; j < 16; ++j) sB[0][lb_r][lb_c + j] = bp[j];
        __syncthreads();
    }
#endif

    for (int kc = 0; kc < nk; ++kc) {
        const int cur = kc & 1;
        const int nxt = cur ^ 1;
        const int k1  = (kc + 1) * BK;

#if HAVE_TDM
        if (tid == 0 && k1 < K) {
            tdm_load_2d_bf16(A + (size_t)rowBase * K + k1,
                             (unsigned)(size_t)&sA[nxt][0][0],
                             (unsigned)K, (unsigned)M, BK, BM, (unsigned long)K, 3u);
            tdm_load_2d_bf16(Bm + (size_t)k1 * N + colBase,
                             (unsigned)(size_t)&sB[nxt][0][0],
                             (unsigned)N, (unsigned)K, BN, BK, (unsigned long)N, 5u);
        }
#endif

        // assemble fragments per CDNA5 16-bit layouts, run 8 WMMAs
        v16bf afrag[4], bfrag[2];
        #pragma unroll
        for (int i = 0; i < 4; ++i) {
            v16us t;
            #pragma unroll
            for (int j = 0; j < 16; ++j) {
                const int kk = (j & 7) + (half << 3) + ((j >> 3) << 4);
                t[j] = sA[cur][waveRow + i * 16 + mn][kk];
            }
            afrag[i] = __builtin_bit_cast(v16bf, t);
        }
        #pragma unroll
        for (int i = 0; i < 2; ++i) {
            v16us t;
            #pragma unroll
            for (int j = 0; j < 16; ++j) {
                const int kk = (j & 7) + (half << 3) + ((j >> 3) << 4);
                t[j] = sB[cur][kk][waveCol + i * 16 + mn];
            }
            bfrag[i] = __builtin_bit_cast(v16bf, t);
        }
        #pragma unroll
        for (int i = 0; i < 4; ++i)
            #pragma unroll
            for (int j = 0; j < 2; ++j)
                acc[i][j] = __builtin_amdgcn_wmma_f32_16x16x32_bf16(
                    false, afrag[i], false, bfrag[j],
                    (short)0, acc[i][j], false, false);

#if HAVE_TDM
        if (tid == 0 && k1 < K) wait_tensorcnt0();
        __syncthreads();
#else
        __syncthreads();
        if (k1 < K) {
            const int la_r = tid >> 1, la_c = (tid & 1) * 16;
            const int lb_r = tid >> 3, lb_c = (tid & 7) * 16;
            const unsigned short* ap = A + (size_t)(rowBase + la_r) * K + k1 + la_c;
            const unsigned short* bp = Bm + (size_t)(k1 + lb_r) * N + colBase + lb_c;
            #pragma unroll
            for (int j = 0; j < 16; ++j) sA[nxt][la_r][la_c + j] = ap[j];
            #pragma unroll
            for (int j = 0; j < 16; ++j) sB[nxt][lb_r][lb_c + j] = bp[j];
            __syncthreads();
        }
#endif
    }

    // store: VGPR r -> M = r + 8*half ; N = mn
    #pragma unroll
    for (int i = 0; i < 4; ++i)
        #pragma unroll
        for (int j = 0; j < 2; ++j) {
            const int col = colBase + waveCol + j * 16 + mn;
            const int row0 = rowBase + waveRow + i * 16 + half * 8;
            #pragma unroll
            for (int r = 0; r < 8; ++r)
                C[(size_t)(row0 + r) * N + col] = acc[i][j][r];
        }
}

// ---------------------------------------------------------------------------
// small GEMM: ba(B*S, 64) = x(B*S, H) @ w_ba(H, 64)
// ---------------------------------------------------------------------------
__global__ __launch_bounds__(256) void ba_gemm(
    const float* __restrict__ x, const float* __restrict__ w_ba,
    float* __restrict__ ba)
{
    const int idx = blockIdx.x * 256 + threadIdx.x;
    const int col = idx & 63;
    const int row = idx >> 6;
    const float* xr = x + (size_t)row * H_;
    float acc = 0.f;
    for (int k = 0; k < H_; ++k) acc += xr[k] * w_ba[k * FBA_ + col];
    ba[idx] = acc;
}

// ---------------------------------------------------------------------------
// causal depthwise conv (K=4) + bias + SiLU, gathering channels from qkvz
// ---------------------------------------------------------------------------
__global__ __launch_bounds__(256) void conv_silu_kernel(
    const float* __restrict__ qkvz, const float* __restrict__ conv_w,
    const float* __restrict__ conv_b, float* __restrict__ mixed, int total)
{
    const int idx = blockIdx.x * 256 + threadIdx.x;
    if (idx >= total) return;
    const int c  = idx % CD_;
    const int bs = idx / CD_;
    const int s  = bs % S_;
    const int b  = bs / S_;

    int off;
    if (c < HK_ * DK_) {
        off = (c >> 7) * 768 + (c & 127);
    } else if (c < 2 * HK_ * DK_) {
        const int cc = c - HK_ * DK_;
        off = (cc >> 7) * 768 + 128 + (cc & 127);
    } else {
        const int cv = c - 2 * HK_ * DK_;
        const int hv = cv >> 7, d = cv & 127;
        off = (hv >> 1) * 768 + 256 + (hv & 1) * 128 + d;
    }

    float acc = conv_b[c];
    #pragma unroll
    for (int j = 0; j < K_; ++j) {
        const int tt = s - (K_ - 1) + j;
        if (tt >= 0)
            acc += conv_w[c * K_ + j] * qkvz[(size_t)(b * S_ + tt) * FQKVZ_ + off];
    }
    acc = acc / (1.f + __expf(-acc));
    mixed[idx] = acc;
}

// ---------------------------------------------------------------------------
// per-head l2norm of q and k (in place); q additionally scaled by DK^-0.5
// ---------------------------------------------------------------------------
__global__ __launch_bounds__(128) void l2norm_kernel(float* __restrict__ mixed)
{
    const int blk = blockIdx.x;
    const int bs  = blk >> 5;
    const int rem = blk & 31;
    const int isK = rem >> 4;
    const int hk  = rem & 15;
    const int tid = threadIdx.x;
    const size_t base = (size_t)bs * CD_ + (size_t)isK * (HK_ * DK_) + hk * 128;

    float v = mixed[base + tid];
    __shared__ float red[128];
    red[tid] = v * v;
    __syncthreads();
    for (int sft = 64; sft > 0; sft >>= 1) {
        if (tid < sft) red[tid] += red[tid + sft];
        __syncthreads();
    }
    float scale = rsqrtf(red[0] + EPS_);
    if (!isK) scale *= 0.08838834764831845f;
    mixed[base + tid] = v * scale;
}

// ---------------------------------------------------------------------------
// decay g = -exp(a_log)*softplus(a + dt_bias), beta = sigmoid(.)
// ---------------------------------------------------------------------------
__global__ __launch_bounds__(256) void gb_kernel(
    const float* __restrict__ ba, const float* __restrict__ a_log,
    const float* __restrict__ dt_bias, float* __restrict__ g,
    float* __restrict__ beta, int total)
{
    const int idx = blockIdx.x * 256 + threadIdx.x;
    if (idx >= total) return;
    const int hv = idx % HV_;
    const int bs = idx / HV_;
    const int hk = hv >> 1, gi = hv & 1;
    const float bsrc = ba[(size_t)bs * FBA_ + hk * 4 + gi];
    const float asrc = ba[(size_t)bs * FBA_ + hk * 4 + G_ + gi];
    beta[idx] = 1.f / (1.f + __expf(-bsrc));
    const float xa = asrc + dt_bias[hv];
    const float sp = (xa > 20.f) ? xa : log1pf(__expf(xa));
    g[idx] = -__expf(a_log[hv]) * sp;
}

// ---------------------------------------------------------------------------
// gated delta-rule scan. One block = (b, hv, 32 state columns).
// 256 threads: 8 lanes per column x 32 columns; 16 state rows per thread.
// ---------------------------------------------------------------------------
__global__ __launch_bounds__(256) void scan_kernel(
    const float* __restrict__ mixed, const float* __restrict__ gbuf,
    const float* __restrict__ bbuf, float* __restrict__ obuf)
{
    const int blk = blockIdx.x;
    const int b   = blk >> 7;
    const int hv  = (blk >> 2) & 31;
    const int cb  = blk & 3;
    const int hk  = hv >> 1;
    const int tid = threadIdx.x;
    const int colg = cb * 32 + (tid >> 3);
    const int r0   = (tid & 7) * 16;

    __shared__ float sk[128], sq[128], sg[2];

    float st[16];
    #pragma unroll
    for (int i = 0; i < 16; ++i) st[i] = 0.f;

    for (int t = 0; t < S_; ++t) {
        const size_t rb = (size_t)(b * S_ + t) * CD_;
        if (tid < 128)       sk[tid]       = mixed[rb + HK_ * DK_ + hk * 128 + tid];
        else                 sq[tid - 128] = mixed[rb + hk * 128 + (tid - 128)];
        if (tid == 0) {
            sg[0] = gbuf[(size_t)(b * S_ + t) * HV_ + hv];
            sg[1] = bbuf[(size_t)(b * S_ + t) * HV_ + hv];
        }
        __syncthreads();

        const float eg = __expf(sg[0]);
        const float bt = sg[1];
        const float vt = mixed[rb + 2 * HK_ * DK_ + hv * 128 + colg];

        float p = 0.f;
        #pragma unroll
        for (int i = 0; i < 16; ++i) { st[i] *= eg; p += sk[r0 + i] * st[i]; }
        p += __shfl_xor(p, 1, 32);
        p += __shfl_xor(p, 2, 32);
        p += __shfl_xor(p, 4, 32);

        const float delta = (vt - p) * bt;

        float o = 0.f;
        #pragma unroll
        for (int i = 0; i < 16; ++i) { st[i] += sk[r0 + i] * delta; o += sq[r0 + i] * st[i]; }
        o += __shfl_xor(o, 1, 32);
        o += __shfl_xor(o, 2, 32);
        o += __shfl_xor(o, 4, 32);

        if ((tid & 7) == 0)
            obuf[((size_t)(b * S_ + t) * HV_ + hv) * DV_ + colg] = o;
        __syncthreads();
    }
}

// ---------------------------------------------------------------------------
// o *= silu(z); o = rmsnorm(o) * norm_w ; emit bf16 for the output GEMM
// ---------------------------------------------------------------------------
__global__ __launch_bounds__(128) void gatenorm_kernel(
    const float* __restrict__ obuf, const float* __restrict__ qkvz,
    const float* __restrict__ norm_w, unsigned short* __restrict__ o_bf16)
{
    const int blk = blockIdx.x;
    const int bs  = blk >> 5;
    const int hv  = blk & 31;
    const int hk  = hv >> 1, gi = hv & 1;
    const int tid = threadIdx.x;
    const size_t ob = ((size_t)bs * HV_ + hv) * DV_ + tid;

    float o = obuf[ob];
    const float z = qkvz[(size_t)bs * FQKVZ_ + hk * 768 + 512 + gi * 128 + tid];
    o *= z / (1.f + __expf(-z));

    __shared__ float red[128];
    red[tid] = o * o;
    __syncthreads();
    for (int sft = 64; sft > 0; sft >>= 1) {
        if (tid < sft) red[tid] += red[tid + sft];
        __syncthreads();
    }
    const float ms = red[0] * (1.f / 128.f);
    o *= rsqrtf(ms + EPS_) * norm_w[tid];
    o_bf16[ob] = f32_to_bf16_rne(o);
}

// ---------------------------------------------------------------------------
extern "C" void kernel_launch(void* const* d_in, const int* in_sizes, int n_in,
                              void* d_out, int out_size, void* d_ws, size_t ws_size,
                              hipStream_t stream)
{
    (void)in_sizes; (void)n_in; (void)out_size; (void)ws_size;
    const float* x       = (const float*)d_in[0];
    const float* w_qkvz  = (const float*)d_in[1];
    const float* w_ba    = (const float*)d_in[2];
    const float* a_log   = (const float*)d_in[3];
    const float* dt_bias = (const float*)d_in[4];
    const float* conv_w  = (const float*)d_in[5];
    const float* conv_b  = (const float*)d_in[6];
    const float* norm_w  = (const float*)d_in[7];
    const float* w_o     = (const float*)d_in[8];
    float* out = (float*)d_out;

    char* ws = (char*)d_ws;
    size_t off = 0;
    auto alloc = [&](size_t bytes) -> void* {
        void* p = ws + off;
        off = (off + bytes + 255) & ~(size_t)255;
        return p;
    };
    const int BS = B_ * S_;
    float*          qkvz  = (float*)alloc((size_t)BS * FQKVZ_ * 4);
    float*          ba    = (float*)alloc((size_t)BS * FBA_ * 4);
    float*          mixed = (float*)alloc((size_t)BS * CD_ * 4);
    float*          gbuf  = (float*)alloc((size_t)BS * HV_ * 4);
    float*          bbuf  = (float*)alloc((size_t)BS * HV_ * 4);
    float*          obuf  = (float*)alloc((size_t)BS * HV_ * DV_ * 4);
    unsigned short* xbf   = (unsigned short*)alloc((size_t)BS * H_ * 2);
    unsigned short* wqbf  = (unsigned short*)alloc((size_t)H_ * FQKVZ_ * 2);
    unsigned short* wobf  = (unsigned short*)alloc((size_t)HV_ * DV_ * H_ * 2);
    unsigned short* obf   = (unsigned short*)alloc((size_t)BS * HV_ * DV_ * 2);

    dim3 blk256(256);

    // 0) bf16 conversions for WMMA GEMM operands
    {
        long n;
        n = (long)BS * H_;
        cvt_bf16_kernel<<<(unsigned)((n + 255) / 256), blk256, 0, stream>>>(x, xbf, n);
        n = (long)H_ * FQKVZ_;
        cvt_bf16_kernel<<<(unsigned)((n + 255) / 256), blk256, 0, stream>>>(w_qkvz, wqbf, n);
        n = (long)HV_ * DV_ * H_;
        cvt_bf16_kernel<<<(unsigned)((n + 255) / 256), blk256, 0, stream>>>(w_o, wobf, n);
    }

    // 1) qkvz = x @ w_qkvz  (WMMA bf16, TDM-staged tiles)
    dim3 g1(FQKVZ_ / 128, BS / 128);
    wmma_gemm_bf16_tdm<<<g1, blk256, 0, stream>>>(xbf, wqbf, qkvz, BS, FQKVZ_, H_);

    // 2) ba = x @ w_ba
    ba_gemm<<<(BS * FBA_) / 256, blk256, 0, stream>>>(x, w_ba, ba);

    // 3) depthwise causal conv + SiLU
    const int totalc = BS * CD_;
    conv_silu_kernel<<<(totalc + 255) / 256, blk256, 0, stream>>>(
        qkvz, conv_w, conv_b, mixed, totalc);

    // 4) per-head l2 norm of q, k
    l2norm_kernel<<<BS * 2 * HK_, 128, 0, stream>>>(mixed);

    // 5) g, beta
    const int totg = BS * HV_;
    gb_kernel<<<(totg + 255) / 256, blk256, 0, stream>>>(
        ba, a_log, dt_bias, gbuf, bbuf, totg);

    // 6) gated delta-rule scan
    scan_kernel<<<B_ * HV_ * 4, blk256, 0, stream>>>(mixed, gbuf, bbuf, obuf);

    // 7) gate with silu(z) + RMS norm -> bf16
    gatenorm_kernel<<<BS * HV_, 128, 0, stream>>>(obuf, qkvz, norm_w, obf);

    // 8) out = o @ w_o  (WMMA bf16, TDM-staged tiles)
    dim3 g2(H_ / 128, BS / 128);
    wmma_gemm_bf16_tdm<<<g2, blk256, 0, stream>>>(obf, wobf, out, BS, H_, HV_ * DV_);
}
